// AttentionLayer_90640989815298
// MI455X (gfx1250) — matrix operand order repaired and checked by reading, not compiled, not run on hardware
//
#include <hip/hip_runtime.h>

// ---------------------------------------------------------------------------
// AttentionLayer for MI455X (gfx1250): f16 WMMA pipeline + async LDS staging
//   B=4, L=2048, D_MODEL=1024, H=16, DK=DV=64
// ---------------------------------------------------------------------------

#define BATCH   4
#define SEQ     2048
#define DMODEL  1024
#define NH      16
#define DH      64          // head dim (DK == DV == 64)

typedef __attribute__((ext_vector_type(16))) _Float16 v16h;
typedef __attribute__((ext_vector_type(8)))  _Float16 v8h;
typedef __attribute__((ext_vector_type(8)))  float    v8f;

__device__ __forceinline__ v8f wmma_f16(v16h a, v16h b, v8f c) {
    // D = A(16x32 f16) * B(32x16 f16) + C(16x16 f32)
    return __builtin_amdgcn_wmma_f32_16x16x32_f16(
        /*neg_a=*/false, a, /*neg_b=*/false, b,
        /*c_mod=*/(short)0, c, /*reuse_a=*/false, /*reuse_b=*/false);
}

// A-matrix fragment (16x32, M = lane%16). rowptr points at row start (+32*j
// for the k-frag). Layout: lanes 0-15 hold k = 0..15,
// lanes 16-31 hold k = 16..31 (matches load_b16).
__device__ __forceinline__ v16h load_a16(const _Float16* rowptr, int lane) {
    int g = (lane >> 4) & 1;
    union { v16h v; v8h h[2]; } u;
    u.h[0] = *(const v8h*)(rowptr + 16 * g);
    u.h[1] = *(const v8h*)(rowptr + 16 * g + 8);
    return u.v;
}

// B-matrix fragment (32x16, N = lane%16). p points at the N-column's k0.
// Layout used here (same as load_a16): lanes 0-15 hold k = 0..15, lanes 16-31 hold k = 16..31,
// contiguous per lane -> single 32B load when column data is k-contiguous.
__device__ __forceinline__ v16h load_b16(const _Float16* p, int lane) {
    return *(const v16h*)(p + 16 * ((lane >> 4) & 1));
}

// Async copy of 16 bytes global -> LDS (per active lane), tracked by ASYNCcnt.
__device__ __forceinline__ void async_g2l_b128(const void* lds_dst,
                                               const void* gaddr) {
    unsigned lds_off = (unsigned)(size_t)lds_dst;           // LDS offset
    unsigned long long ga = (unsigned long long)(size_t)gaddr;
    asm volatile("global_load_async_to_lds_b128 %0, %1, off"
                 :: "v"(lds_off), "v"(ga)
                 : "memory");
}

// ---------------------------------------------------------------------------
// Kernel 1: fused QKV projection.  Y = X @ W + b  (fp32 in, f16 out)
//   grid = (DMODEL/128, B*L/128, 3), block = 256 (8 waves, 2x4 wave grid)
//   k-step 64 (2 WMMA k-frags per stage -> 16 WMMAs/wave between barriers)
//   proj 0 -> Qh [b,h,l,64]   proj 1 -> Kh [b,h,l,64]   proj 2 -> Vt [b,h,d,L]
// ---------------------------------------------------------------------------
__global__ __launch_bounds__(256) void qkv_proj_kernel(
    const float* __restrict__ Xq, const float* __restrict__ Xk,
    const float* __restrict__ Xv,
    const float* __restrict__ Wq, const float* __restrict__ bq,
    const float* __restrict__ Wk, const float* __restrict__ bk,
    const float* __restrict__ Wv, const float* __restrict__ bv,
    _Float16* __restrict__ Qh, _Float16* __restrict__ Kh,
    _Float16* __restrict__ Vt)
{
    const int proj = blockIdx.z;
    const float* X    = (proj == 0) ? Xq : (proj == 1) ? Xk : Xv;
    const float* W    = (proj == 0) ? Wq : (proj == 1) ? Wk : Wv;
    const float* bias = (proj == 0) ? bq : (proj == 1) ? bk : bv;

    __shared__ __align__(32) _Float16 sA[128 * 64];   // [m][k]   16 KB
    __shared__ __align__(32) _Float16 sB[128 * 64];   // [n][k]   16 KB

    const int t    = threadIdx.x;
    const int lane = t & 31;
    const int w    = t >> 5;         // wave 0..7
    const int wm   = w >> 2;         // 0..1  (M)
    const int wn   = w & 3;          // 0..3  (N)
    const int li   = lane & 15;
    const int g    = lane >> 4;
    const int m0   = blockIdx.y * 128;
    const int n0   = blockIdx.x * 128;

    const int mm  = t >> 1;          // A staging row     (0..127)
    const int ko  = (t & 1) * 32;    // A staging k base  {0,32}
    const int kk  = t >> 2;          // B staging k row   (0..63)
    const int nn0 = (t & 3) * 32;    // B staging n base  {0,32,64,96}

    v8f acc[4][2] = {};

    for (int k0 = 0; k0 < DMODEL; k0 += 64) {
        // Stage A tile 128x64: fp32 -> f16, row-major
        {
            const float* src = X + (size_t)(m0 + mm) * DMODEL + k0 + ko;
            v16h t0, t1;
            #pragma unroll
            for (int i = 0; i < 16; ++i) t0[i] = (_Float16)src[i];
            #pragma unroll
            for (int i = 0; i < 16; ++i) t1[i] = (_Float16)src[16 + i];
            *(v16h*)(sA + mm * 64 + ko)      = t0;
            *(v16h*)(sA + mm * 64 + ko + 16) = t1;
            if (k0 + 64 < DMODEL) __builtin_prefetch(src + 64, 0, 3);
        }
        // Stage W tile 64x128 transposed into sB[n][k]
        {
            const float* src = W + (size_t)(k0 + kk) * DMODEL + n0 + nn0;
            #pragma unroll
            for (int i = 0; i < 32; ++i)
                sB[(nn0 + i) * 64 + kk] = (_Float16)src[i];
            if (k0 + 64 < DMODEL)
                __builtin_prefetch(src + (size_t)64 * DMODEL, 0, 3);
        }
        __syncthreads();

        #pragma unroll
        for (int j = 0; j < 2; ++j) {
            v16h af[4];
            #pragma unroll
            for (int mt = 0; mt < 4; ++mt)
                af[mt] = load_a16(sA + (64 * wm + 16 * mt + li) * 64 + 32 * j,
                                  lane);
            #pragma unroll
            for (int nt = 0; nt < 2; ++nt) {
                v16h bf = load_b16(sB + (32 * wn + 16 * nt + li) * 64 + 32 * j,
                                   lane);
                #pragma unroll
                for (int mt = 0; mt < 4; ++mt)
                    acc[mt][nt] = wmma_f16(af[mt], bf, acc[mt][nt]);
            }
        }
        __syncthreads();
    }

    // Epilogue: C-frag element (M = r + 8g, N = li) within each 16x16 tile.
    const int b = m0 >> 11;               // 2048 rows per batch, tiles aligned
    #pragma unroll
    for (int mt = 0; mt < 4; ++mt) {
        int mrow0 = m0 + 64 * wm + 16 * mt + 8 * g;   // 8 consecutive rows
        int lrow0 = mrow0 - b * SEQ;
        #pragma unroll
        for (int nt = 0; nt < 2; ++nt) {
            int n = n0 + 32 * wn + 16 * nt + li;
            int h = n >> 6, d = n & 63;
            float bs = bias[n];
            if (proj < 2) {
                _Float16* dst = ((proj == 0) ? Qh : Kh) +
                    ((size_t)((b * NH + h) * SEQ + lrow0)) * DH + d;
                #pragma unroll
                for (int r = 0; r < 8; ++r)
                    dst[(size_t)r * DH] = (_Float16)(acc[mt][nt][r] + bs);
            } else {
                v8h val;
                #pragma unroll
                for (int r = 0; r < 8; ++r)
                    val[r] = (_Float16)(acc[mt][nt][r] + bs);
                *(v8h*)(Vt + ((size_t)(b * NH + h) * DH + d) * SEQ + lrow0) = val;
            }
        }
    }
}

// ---------------------------------------------------------------------------
// Kernel 2: causal flash attention per (query-block of 64, b*h).
//   grid = (L/64, B*H), block = 128 (4 waves, each owns 16 query rows)
//   K/V tiles double-buffered in LDS via async global->LDS copies (ASYNCcnt).
// ---------------------------------------------------------------------------
__global__ __launch_bounds__(128) void attn_kernel(
    const _Float16* __restrict__ Qh, const _Float16* __restrict__ Kh,
    const _Float16* __restrict__ Vt, _Float16* __restrict__ Ob)
{
    __shared__ __align__(16) _Float16 sK[2][64 * DH];  // [key][dh] per buffer
    __shared__ __align__(16) _Float16 sV[2][64 * DH];  // [dv][s]   per buffer
    __shared__ __align__(16) _Float16 pb[4][16 * 64];  // per-wave P tile

    const int t    = threadIdx.x;
    const int lane = t & 31;
    const int w    = t >> 5;
    const int li   = lane & 15;
    const int g    = lane >> 4;
    const int bh   = blockIdx.y;
    const int b    = bh >> 4;
    const int h    = bh & 15;
    const int q0   = blockIdx.x * 64;
    const int qrow0 = q0 + w * 16;
    const float scale = 0.125f;   // 1/sqrt(64)

    const _Float16* Qp = Qh + (size_t)bh * SEQ * DH;
    const _Float16* Kp = Kh + (size_t)bh * SEQ * DH;
    const _Float16* Vp = Vt + (size_t)bh * DH * SEQ;

    v16h qf[2];
    #pragma unroll
    for (int j = 0; j < 2; ++j)
        qf[j] = load_a16(Qp + (size_t)(qrow0 + li) * DH + 32 * j, lane);

    v8f o[4] = {};
    float m[8], lsum[8];
    #pragma unroll
    for (int r = 0; r < 8; ++r) { m[r] = -1e30f; lsum[r] = 0.f; }

    const int nblk = (q0 >> 6) + 1;

    // Prefetch block 0 into buffer 0 (K tile 8KB + V tile 8KB, 16B chunks,
    // 8 async ops per thread -> 8 per wave outstanding per block).
    #pragma unroll
    for (int i = 0; i < 4; ++i) {
        int c   = t + i * 128;        // 0..511
        int row = c >> 3;
        int off = (c & 7) * 8;
        async_g2l_b128(&sK[0][row * DH + off], Kp + (size_t)row * DH + off);
        async_g2l_b128(&sV[0][row * DH + off], Vp + (size_t)row * SEQ + off);
    }

    for (int blk = 0; blk < nblk; ++blk) {
        const int s0  = blk * 64;
        const int cur = blk & 1;

        if (blk + 1 < nblk) {
            const int s1  = s0 + 64;
            const int nxt = cur ^ 1;
            #pragma unroll
            for (int i = 0; i < 4; ++i) {
                int c   = t + i * 128;
                int row = c >> 3;
                int off = (c & 7) * 8;
                async_g2l_b128(&sK[nxt][row * DH + off],
                               Kp + (size_t)(s1 + row) * DH + off);
                async_g2l_b128(&sV[nxt][row * DH + off],
                               Vp + (size_t)row * SEQ + s1 + off);
            }
            asm volatile("s_wait_asynccnt 8" ::: "memory");
        } else {
            asm volatile("s_wait_asynccnt 0" ::: "memory");
        }
        __syncthreads();   // block data in LDS; prior pb loads also done

        // ---- S = Q @ K^T, 16x64 in 4 f32 frags ----
        v8f s[4];
        #pragma unroll
        for (int nt = 0; nt < 4; ++nt) {
            v8f c = {};
            #pragma unroll
            for (int j = 0; j < 2; ++j) {
                v16h kf = load_b16(&sK[cur][(16 * nt + li) * DH + 32 * j],
                                   lane);
                c = wmma_f16(qf[j], kf, c);
            }
            s[nt] = c;
        }
        // ---- causal mask + block row max ----
        float bm[8];
        #pragma unroll
        for (int r = 0; r < 8; ++r) bm[r] = -1e30f;
        #pragma unroll
        for (int nt = 0; nt < 4; ++nt) {
            int col = s0 + 16 * nt + li;
            #pragma unroll
            for (int r = 0; r < 8; ++r) {
                int row = qrow0 + r + 8 * g;
                float sv = (col <= row) ? s[nt][r] : -1e30f;
                s[nt][r] = sv;
                bm[r] = fmaxf(bm[r], sv);
            }
        }
        #pragma unroll
        for (int r = 0; r < 8; ++r)
            #pragma unroll
            for (int off = 1; off < 16; off <<= 1)
                bm[r] = fmaxf(bm[r], __shfl_xor(bm[r], off, 16));

        float alpha[8], ps[8];
        #pragma unroll
        for (int r = 0; r < 8; ++r) {
            float nm = fmaxf(m[r], bm[r]);
            alpha[r] = __expf((m[r] - nm) * scale);
            m[r] = nm;
            ps[r] = 0.f;
        }

        // ---- P = exp(scale*(S-m)) -> per-wave LDS tile [16][64] ----
        #pragma unroll
        for (int nt = 0; nt < 4; ++nt) {
            #pragma unroll
            for (int r = 0; r < 8; ++r) {
                float p = __expf((s[nt][r] - m[r]) * scale);
                ps[r] += p;
                pb[w][(r + 8 * g) * 64 + 16 * nt + li] = (_Float16)p;
            }
        }
        #pragma unroll
        for (int r = 0; r < 8; ++r) {
            #pragma unroll
            for (int off = 1; off < 16; off <<= 1)
                ps[r] += __shfl_xor(ps[r], off, 16);
            lsum[r] = lsum[r] * alpha[r] + ps[r];
        }
        // rescale running O
        #pragma unroll
        for (int nt2 = 0; nt2 < 4; ++nt2)
            #pragma unroll
            for (int r = 0; r < 8; ++r) o[nt2][r] *= alpha[r];
        __syncthreads();   // P visible in LDS

        // ---- O += P @ V ----
        v16h pf[2];
        #pragma unroll
        for (int j = 0; j < 2; ++j)
            pf[j] = load_a16(&pb[w][li * 64] + 32 * j, lane);
        #pragma unroll
        for (int nt2 = 0; nt2 < 4; ++nt2) {
            #pragma unroll
            for (int j = 0; j < 2; ++j) {
                v16h vf = load_b16(&sV[cur][(16 * nt2 + li) * DH + 32 * j],
                                   lane);
                o[nt2] = wmma_f16(pf[j], vf, o[nt2]);
            }
        }
        __syncthreads();   // sK/sV/pb reads done before next prefetch lands
    }

    // ---- epilogue: Ob[b, row, h*64 + dv] = O / lsum  (f16) ----
    #pragma unroll
    for (int nt2 = 0; nt2 < 4; ++nt2) {
        int dv = h * DH + 16 * nt2 + li;
        #pragma unroll
        for (int r = 0; r < 8; ++r) {
            int row = qrow0 + r + 8 * g;
            Ob[(size_t)(b * SEQ + row) * DMODEL + dv] =
                (_Float16)(o[nt2][r] / lsum[r]);
        }
    }
}

// ---------------------------------------------------------------------------
// Kernel 3: output projection  Y = Ob(f16) @ Wo + bo  (fp32 out)
//   grid = (DMODEL/128, B*L/128), block = 256, k-step 64
// ---------------------------------------------------------------------------
__global__ __launch_bounds__(256) void out_proj_kernel(
    const _Float16* __restrict__ Ob, const float* __restrict__ Wo,
    const float* __restrict__ bo, float* __restrict__ Y)
{
    __shared__ __align__(32) _Float16 sB[128 * 64];   // [n][k]  16 KB

    const int t    = threadIdx.x;
    const int lane = t & 31;
    const int w    = t >> 5;
    const int wm   = w >> 2;
    const int wn   = w & 3;
    const int li   = lane & 15;
    const int g    = lane >> 4;
    const int m0   = blockIdx.y * 128;
    const int n0   = blockIdx.x * 128;

    const int kk  = t >> 2;          // 0..63
    const int nn0 = (t & 3) * 32;    // {0,32,64,96}

    v8f acc[4][2] = {};

    for (int k0 = 0; k0 < DMODEL; k0 += 64) {
        {
            const float* src = Wo + (size_t)(k0 + kk) * DMODEL + n0 + nn0;
            #pragma unroll
            for (int i = 0; i < 32; ++i)
                sB[(nn0 + i) * 64 + kk] = (_Float16)src[i];
            if (k0 + 64 < DMODEL)
                __builtin_prefetch(src + (size_t)64 * DMODEL, 0, 3);
        }
        __syncthreads();

        #pragma unroll
        for (int j = 0; j < 2; ++j) {
            v16h af[4];
            #pragma unroll
            for (int mt = 0; mt < 4; ++mt)
                af[mt] = load_a16(
                    Ob + (size_t)(m0 + 64 * wm + 16 * mt + li) * DMODEL + k0 +
                        32 * j, lane);
            #pragma unroll
            for (int nt = 0; nt < 2; ++nt) {
                v16h bf = load_b16(sB + (32 * wn + 16 * nt + li) * 64 + 32 * j,
                                   lane);
                #pragma unroll
                for (int mt = 0; mt < 4; ++mt)
                    acc[mt][nt] = wmma_f16(af[mt], bf, acc[mt][nt]);
            }
        }
        __syncthreads();
    }

    #pragma unroll
    for (int mt = 0; mt < 4; ++mt) {
        int mrow0 = m0 + 64 * wm + 16 * mt + 8 * g;
        #pragma unroll
        for (int nt = 0; nt < 2; ++nt) {
            int n = n0 + 32 * wn + 16 * nt + li;
            float bs = bo[n];
            #pragma unroll
            for (int r = 0; r < 8; ++r)
                Y[(size_t)(mrow0 + r) * DMODEL + n] = acc[mt][nt][r] + bs;
        }
    }
}

// ---------------------------------------------------------------------------
extern "C" void kernel_launch(void* const* d_in, const int* in_sizes, int n_in,
                              void* d_out, int out_size, void* d_ws,
                              size_t ws_size, hipStream_t stream)
{
    (void)in_sizes; (void)n_in; (void)out_size; (void)ws_size;

    const float* queries = (const float*)d_in[0];
    const float* keys    = (const float*)d_in[1];
    const float* values  = (const float*)d_in[2];
    const float* Wq = (const float*)d_in[3];
    const float* bq = (const float*)d_in[4];
    const float* Wk = (const float*)d_in[5];
    const float* bk = (const float*)d_in[6];
    const float* Wv = (const float*)d_in[7];
    const float* bv = (const float*)d_in[8];
    const float* Wo = (const float*)d_in[9];
    const float* bo = (const float*)d_in[10];
    float* out = (float*)d_out;

    // Workspace: 4 x (B*H*L*64) f16 buffers = 64 MB total
    const size_t HBUF = (size_t)BATCH * NH * SEQ * DH;
    _Float16* Qh = (_Float16*)d_ws;
    _Float16* Kh = Qh + HBUF;
    _Float16* Vt = Kh + HBUF;
    _Float16* Ob = Vt + HBUF;

    qkv_proj_kernel<<<dim3(DMODEL / 128, (BATCH * SEQ) / 128, 3), 256, 0,
                      stream>>>(queries, keys, values, Wq, bq, Wk, bk, Wv, bv,
                                Qh, Kh, Vt);

    attn_kernel<<<dim3(SEQ / 64, BATCH * NH), 128, 0, stream>>>(Qh, Kh, Vt,
                                                                Ob);

    out_proj_kernel<<<dim3(DMODEL / 128, (BATCH * SEQ) / 128), 256, 0,
                      stream>>>(Ob, Wo, bo, out);
}
